// SentimentClassifier_40295383171778
// MI455X (gfx1250) — compile-verified, matrix operand order checked by reading, not run on hardware
//
#include <hip/hip_runtime.h>

typedef __attribute__((ext_vector_type(16))) __bf16 v16bf;
typedef __attribute__((ext_vector_type(8)))  __bf16 v8bf;
typedef __attribute__((ext_vector_type(8)))  float  v8f;

#define EDIM 256
#define HDIM 256
#define G4H  1024
#define BATCH 64
#define TSEQ 512

union abf16 { v16bf v; v8bf h[2]; };

static __device__ __forceinline__ unsigned short f2bf_bits(float f) {
    return __builtin_bit_cast(unsigned short, (__bf16)f);
}
static __device__ __forceinline__ float sigf(float x) {
    return 1.0f / (1.0f + __expf(-x));
}

// ---------------------------------------------------------------------------
// One-time fp32 -> bf16 conversion of the embedding table (row-major [V, E]).
// ---------------------------------------------------------------------------
__global__ void emb_to_bf16(const float* __restrict__ emb,
                            unsigned short* __restrict__ out, int n) {
    int i = (blockIdx.x * blockDim.x + threadIdx.x) * 4;
    if (i < n) {
        float4 v = *(const float4*)(emb + i);
        ushort4 o;
        o.x = f2bf_bits(v.x); o.y = f2bf_bits(v.y);
        o.z = f2bf_bits(v.z); o.w = f2bf_bits(v.w);
        *(ushort4*)(out + i) = o;
    }
}

// ---------------------------------------------------------------------------
// Pack W [1024, 256] fp32 -> bf16 tiles in the WMMA B-matrix per-lane layout.
// Tile (nt, kt): lane l holds column n = nt*16 + (l&15); element e ->
// K = kt*32 + (l>>4)*8 + (e/8)*16 + (e&7)  (ISA 16-bit half-lane K split).
// ---------------------------------------------------------------------------
__global__ void pack_w_bf16(const float* __restrict__ W,
                            unsigned short* __restrict__ out) {
    int idx    = blockIdx.x * blockDim.x + threadIdx.x;   // 0..262143
    int tile   = idx >> 9;
    int within = idx & 511;
    int lane   = within >> 4;
    int e      = within & 15;
    int kt     = tile & 7;     // 256/32 = 8 K-tiles
    int nt     = tile >> 3;    // 1024/16 = 64 N-tiles
    int n = nt * 16 + (lane & 15);
    int k = kt * 32 + ((lane >> 4) << 3) + ((e >> 3) << 4) + (e & 7);
    out[idx] = f2bf_bits(W[n * EDIM + k]);
}

__global__ void combine_bias(const float* __restrict__ a,
                             const float* __restrict__ b,
                             float* __restrict__ out) {
    int i = blockIdx.x * blockDim.x + threadIdx.x;
    if (i < G4H) out[i] = a[i] + b[i];
}

// ---------------------------------------------------------------------------
// Embedding gather fused with xproj GEMM:  out[r, :] = emb[x[r]] @ W^T + bias
// One wave per 16x64 output strip (4 N-tiles share one A fetch per K-step).
// Token-major rows r = t*64 + b; x is [B, T] int64. A comes from the
// pre-converted bf16 embedding table -> pure b128 loads, no conversion VALU.
// ---------------------------------------------------------------------------
__global__ void xproj_wmma(const long long* __restrict__ x,
                           const unsigned short* __restrict__ ebf,
                           const unsigned short* __restrict__ wpk,
                           const float* __restrict__ bias,
                           float* __restrict__ out,
                           int base_row) {
    int wid  = (blockIdx.x * blockDim.x + threadIdx.x) >> 5;
    int lane = threadIdx.x & 31;
    int mt   = wid >> 4;          // M-tile
    int ntb  = (wid & 15) * 4;    // first of 4 N-tiles
    int half = lane >> 4;
    int nl   = lane & 15;

    // A: 16 token rows, this lane owns row (l&15)
    int row = base_row + mt * 16 + nl;
    int t = row >> 6, b = row & 63;
    int id = (int)x[(size_t)b * TSEQ + t];
    const unsigned short* er = ebf + (size_t)id * EDIM;

    const v16bf* bbase = (const v16bf*)wpk;
    v8f acc[4] = {};
#pragma unroll
    for (int kt = 0; kt < 8; ++kt) {
        int kb = kt * 32 + (half << 3);
        abf16 a;
        a.h[0] = *(const v8bf*)(er + kb);
        a.h[1] = *(const v8bf*)(er + kb + 16);
#pragma unroll
        for (int q = 0; q < 4; ++q) {
            v16bf bm = bbase[((ntb + q) * 8 + kt) * 32 + lane];
            acc[q] = __builtin_amdgcn_wmma_f32_16x16x32_bf16(
                false, a.v, false, bm, (short)0, acc[q], false, false);
        }
    }

    int r0 = mt * 16 + (half << 3);     // C layout: vgpr j -> row j + 8*half
#pragma unroll
    for (int q = 0; q < 4; ++q) {
        int col = (ntb + q) * 16 + nl;
        float bv = bias[col];
#pragma unroll
        for (int j = 0; j < 8; ++j)
            out[(size_t)(r0 + j) * G4H + col] = acc[q][j] + bv;
    }
}

// ---------------------------------------------------------------------------
// Sequential forward LSTM: 1 workgroup, 16 waves, 512 steps.
// h double-buffered in LDS as bf16 (64 KB); c in registers (accumulator
// layout); Whh bf16 re-streamed from L2 each step. Wave w owns h-column
// block w (16 cols) across all 4 gates and all 4 M-tiles, and one LDS A
// fetch per K-step feeds all 4 gate WMMAs. Next step's xproj slice is
// prefetched behind the current step's compute.
// ---------------------------------------------------------------------------
__global__ __launch_bounds__(512, 1)
void lstm_fwd(const float* __restrict__ xproj,
              const unsigned short* __restrict__ whh_pk,
              float* __restrict__ hf_out) {
    __shared__ unsigned short hbuf[2][BATCH * HDIM];   // 64 KB
    int tid  = threadIdx.x;
    int cb   = tid >> 5;          // wave id == column block 0..15
    int lane = tid & 31;
    int half = lane >> 4;
    int nl   = lane & 15;

    for (int i = tid; i < BATCH * HDIM; i += 512) {
        hbuf[0][i] = 0; hbuf[1][i] = 0;
    }
    __syncthreads();

    v8f cst[4] = {};
    const v16bf* bbase = (const v16bf*)whh_pk;

    for (int t = 0; t < TSEQ; ++t) {
        int cur = t & 1, nxt = cur ^ 1;
        const float* zrow = xproj + (size_t)t * BATCH * G4H;
        if (t + 1 < TSEQ) {   // pull next step's z slice toward the WGP
            const float* znext = zrow + (size_t)BATCH * G4H;
#pragma unroll
            for (int p = 0; p < 4; ++p)
                __builtin_prefetch(znext + (size_t)tid * 128 + p * 32, 0, 0);
        }
#pragma unroll
        for (int mt = 0; mt < 4; ++mt) {
            int arow = mt * 16 + nl;    // A row for this lane
            v8f acc[4];
#pragma unroll
            for (int gate = 0; gate < 4; ++gate) {
                int col = gate * HDIM + cb * 16 + nl;
                const float* zp = zrow + (size_t)(mt * 16 + (half << 3)) * G4H + col;
#pragma unroll
                for (int j = 0; j < 8; ++j) acc[gate][j] = zp[(size_t)j * G4H];
            }
#pragma unroll
            for (int kt = 0; kt < 8; ++kt) {
                int kb = kt * 32 + (half << 3);
                abf16 am;
                am.h[0] = *(const v8bf*)&hbuf[cur][arow * HDIM + kb];
                am.h[1] = *(const v8bf*)&hbuf[cur][arow * HDIM + kb + 16];
#pragma unroll
                for (int gate = 0; gate < 4; ++gate) {
                    v16bf bm = bbase[((gate * 16 + cb) * 8 + kt) * 32 + lane];
                    acc[gate] = __builtin_amdgcn_wmma_f32_16x16x32_bf16(
                        false, am.v, false, bm, (short)0, acc[gate], false, false);
                }
            }
            // gate fusion: i,f,g,o -> c,h  (register-local in C layout)
#pragma unroll
            for (int j = 0; j < 8; ++j) {
                float iv = sigf(acc[0][j]);
                float fv = sigf(acc[1][j]);
                float gv = tanhf(acc[2][j]);
                float ov = sigf(acc[3][j]);
                float c  = fv * cst[mt][j] + iv * gv;
                cst[mt][j] = c;
                float h  = ov * tanhf(c);
                int rowm = mt * 16 + j + (half << 3);
                int colh = cb * 16 + nl;
                hbuf[nxt][rowm * HDIM + colh] = f2bf_bits(h);
                if (t == TSEQ - 1) hf_out[rowm * HDIM + colh] = h;
            }
        }
        __syncthreads();
    }
}

// ---------------------------------------------------------------------------
// Backward single step (h0=c0=0 => z = xproj_b) + FC head.
// ---------------------------------------------------------------------------
__global__ void tail_kernel(const float* __restrict__ hf,
                            const float* __restrict__ xb,    // [64, 1024]
                            const float* __restrict__ fc1w,  // [24, 512]
                            const float* __restrict__ fc1b,
                            const float* __restrict__ fc2w,  // [2, 24]
                            const float* __restrict__ fc2b,
                            float* __restrict__ out) {
    __shared__ float hb[BATCH][HDIM];                         // 64 KB
    int tid = threadIdx.x;
    for (int idx = tid; idx < BATCH * HDIM; idx += 256) {
        int r = idx >> 8, j = idx & 255;
        const float* z = xb + (size_t)r * G4H;
        float c = sigf(z[j]) * tanhf(z[2 * HDIM + j]);        // sig(f)*c0 = 0
        hb[r][j] = sigf(z[3 * HDIM + j]) * tanhf(c);
    }
    __syncthreads();
    if (tid < 128) {
        int r = tid >> 1, oo = tid & 1;
        float acc2 = fc2b[oo];
        for (int k = 0; k < 24; ++k) {
            float d = fc1b[k];
            const float* w1 = fc1w + (size_t)k * (2 * HDIM);
            const float* hr = hf + (size_t)r * HDIM;
#pragma unroll 4
            for (int e = 0; e < HDIM; ++e) d += hr[e] * w1[e];
#pragma unroll 4
            for (int e = 0; e < HDIM; ++e) d += hb[r][e] * w1[HDIM + e];
            acc2 += fmaxf(d, 0.f) * fc2w[(size_t)oo * 24 + k];
        }
        out[r * 2 + oo] = acc2;
    }
}

extern "C" void kernel_launch(void* const* d_in, const int* in_sizes, int n_in,
                              void* d_out, int out_size, void* d_ws, size_t ws_size,
                              hipStream_t stream) {
    (void)in_sizes; (void)n_in; (void)out_size; (void)ws_size;
    const long long* x  = (const long long*)d_in[0];
    const float* emb    = (const float*)d_in[1];
    const float* wih_f  = (const float*)d_in[2];
    const float* whh_f  = (const float*)d_in[3];
    const float* bih_f  = (const float*)d_in[4];
    const float* bhh_f  = (const float*)d_in[5];
    const float* wih_b  = (const float*)d_in[6];
    // d_in[7] (w_hh_b) unused: backward direction runs a single step from h0=0
    const float* bih_b  = (const float*)d_in[8];
    const float* bhh_b  = (const float*)d_in[9];
    const float* fc1w   = (const float*)d_in[10];
    const float* fc1b   = (const float*)d_in[11];
    const float* fc2w   = (const float*)d_in[12];
    const float* fc2b   = (const float*)d_in[13];
    float* out = (float*)d_out;

    char* ws = (char*)d_ws;
    unsigned short* wihf_pk = (unsigned short*)(ws + 0);         // 512 KB
    unsigned short* whhf_pk = (unsigned short*)(ws + 524288);    // 512 KB
    unsigned short* wihb_pk = (unsigned short*)(ws + 1048576);   // 512 KB
    float* bias_f = (float*)(ws + 1572864);                      // 4 KB
    float* bias_b = (float*)(ws + 1576960);                      // 4 KB
    float* hf     = (float*)(ws + 1581056);                      // 64 KB
    float* xpb    = (float*)(ws + 1646592);                      // 256 KB
    unsigned short* embbf = (unsigned short*)(ws + 1908736);     // 25.6 MB
    float* xpf    = (float*)(ws + 27508736);                     // 128 MB

    const int nEmb = 50000 * EDIM;                               // 12.8 M elems
    emb_to_bf16<<<(nEmb / 4 + 255) / 256, 256, 0, stream>>>(emb, embbf, nEmb);
    pack_w_bf16<<<1024, 256, 0, stream>>>(wih_f, wihf_pk);
    pack_w_bf16<<<1024, 256, 0, stream>>>(whh_f, whhf_pk);
    pack_w_bf16<<<1024, 256, 0, stream>>>(wih_b, wihb_pk);
    combine_bias<<<4, 256, 0, stream>>>(bih_f, bhh_f, bias_f);
    combine_bias<<<4, 256, 0, stream>>>(bih_b, bhh_b, bias_b);

    // forward xproj: 2048 M-tiles x 16 N-groups, 8 waves/block
    xproj_wmma<<<4096, 256, 0, stream>>>(x, embbf, wihf_pk, bias_f, xpf, 0);
    // backward xproj: only t = T-1 (rows 32704..32767), 4 M-tiles
    xproj_wmma<<<8, 256, 0, stream>>>(x, embbf, wihb_pk, bias_b, xpb, 32704);

    lstm_fwd<<<1, 512, 0, stream>>>(xpf, whhf_pk, hf);
    tail_kernel<<<1, 256, 0, stream>>>(hf, xpb, fc1w, fc1b, fc2w, fc2b, out);
}
